// Top2Gating_26276609917521
// MI455X (gfx1250) — matrix operand order repaired and checked by reading, not compiled, not run on hardware
//
#include <hip/hip_runtime.h>
#include <hip/hip_bf16.h>

typedef __attribute__((ext_vector_type(16))) __bf16          v16bf;
typedef __attribute__((ext_vector_type(16))) unsigned short  v16u;
typedef __attribute__((ext_vector_type(8)))  float           v8f;

#define D_DIM 2048
#define E_DIM 64
#define EPS_GATE 1e-9f
#define NEG_BIG (-3.402823466e+38f)

// ---------------------------------------------------------------------------
// Pre-pass: split fp32 W into bf16 hi/lo (truncation split: w = hi + lo + eps)
// ---------------------------------------------------------------------------
__global__ void split_w_kernel(const float* __restrict__ w,
                               unsigned short* __restrict__ whi,
                               unsigned short* __restrict__ wlo, int n) {
  int i = blockIdx.x * blockDim.x + threadIdx.x;
  if (i >= n) return;
  float f = w[i];
  unsigned int b = __float_as_uint(f);
  float fh = __uint_as_float(b & 0xFFFF0000u);
  float fl = f - fh;
  whi[i] = (unsigned short)(b >> 16);
  wlo[i] = (unsigned short)(__float_as_uint(fl) >> 16);
}

// ---------------------------------------------------------------------------
// Fused gating kernel: logits = x * W^T (bf16 hi/lo 3-term WMMA),
// then per-token softmax + top-2 + combine weights.
// Block = 256 threads = 8 wave32, tile = 128 tokens x 64 experts.
// ---------------------------------------------------------------------------
__global__ __launch_bounds__(256) void gating_kernel(
    const float* __restrict__ x,
    const unsigned short* __restrict__ whi,
    const unsigned short* __restrict__ wlo,
    float* __restrict__ out_cw,   // [n_tokens, 2]
    int* __restrict__ out_idx,    // [n_tokens, 2]
    int n_tokens) {
  __shared__ float lds[128 * 65];   // 65-stride: bank-conflict-free epilogue

  const int lane  = threadIdx.x & 31;
  const int wave  = threadIdx.x >> 5;
  const int m     = lane & 15;          // row (A) / col (B) within tile
  const int khalf = lane >> 4;          // 0 or 1

  int row = blockIdx.x * 128 + wave * 16 + m;
  if (row >= n_tokens) row = n_tokens - 1;    // clamp (tokens % 128 == 0 normally)
  const float* xrow = x + (size_t)row * D_DIM;

  v8f acc[4] = {};   // 4 e-tiles of 16x16 f32

  for (int k0 = 0; k0 < D_DIM; k0 += 32) {
    // ---- A fragment (16x32 bf16): per-lane K = {ka..ka+7, ka+16..ka+23} ----
    const int ka = k0 + khalf * 8;
    v16u ah, al;
#pragma unroll
    for (int j = 0; j < 8; ++j) {
      float f = xrow[ka + j];
      unsigned int b = __float_as_uint(f);
      ah[j] = (unsigned short)(b >> 16);
      float fl = f - __uint_as_float(b & 0xFFFF0000u);
      al[j] = (unsigned short)(__float_as_uint(fl) >> 16);

      float g = xrow[ka + 16 + j];
      unsigned int c = __float_as_uint(g);
      ah[8 + j] = (unsigned short)(c >> 16);
      float gl = g - __uint_as_float(c & 0xFFFF0000u);
      al[8 + j] = (unsigned short)(__float_as_uint(gl) >> 16);
    }
    const v16bf Ah = __builtin_bit_cast(v16bf, ah);
    const v16bf Al = __builtin_bit_cast(v16bf, al);

    // ---- B fragments (32x16 bf16): per-lane K = {kb..kb+7, kb+16..kb+23} ----
    const int kb = k0 + khalf * 8;
#pragma unroll
    for (int et = 0; et < 4; ++et) {
      const size_t woff = (size_t)(et * 16 + m) * D_DIM + kb;
      const unsigned short* ph = whi + woff;
      const unsigned short* pl = wlo + woff;
      v16u bh, bl;
#pragma unroll
      for (int j = 0; j < 8; ++j) { bh[j] = ph[j]; bl[j] = pl[j]; bh[8 + j] = ph[16 + j]; bl[8 + j] = pl[16 + j]; }
      const v16bf Bh = __builtin_bit_cast(v16bf, bh);
      const v16bf Bl = __builtin_bit_cast(v16bf, bl);

      // logits ~= xh*wh + xh*wl + xl*wh  (3-term split, ~fp32 accuracy)
      acc[et] = __builtin_amdgcn_wmma_f32_16x16x32_bf16(
          false, Ah, false, Bh, (short)0, acc[et], false, false);
      acc[et] = __builtin_amdgcn_wmma_f32_16x16x32_bf16(
          false, Ah, false, Bl, (short)0, acc[et], false, false);
      acc[et] = __builtin_amdgcn_wmma_f32_16x16x32_bf16(
          false, Al, false, Bh, (short)0, acc[et], false, false);
    }
  }

  // ---- spill accumulators to LDS ----
  // C/D layout: VGPR r, lanes 0-15 -> M=r, lanes 16-31 -> M=r+8, N = lane&15
  const int lrow = wave * 16 + khalf * 8;
#pragma unroll
  for (int et = 0; et < 4; ++et)
#pragma unroll
    for (int r = 0; r < 8; ++r)
      lds[(lrow + r) * 65 + et * 16 + m] = acc[et][r];

  __syncthreads();

  // ---- epilogue: 128 threads, one token each ----
  const int tid = threadIdx.x;
  if (tid < 128) {
    const int token = blockIdx.x * 128 + tid;
    if (token < n_tokens) {
      const float* lrowp = &lds[tid * 65];
      float v0 = NEG_BIG, v1 = NEG_BIG;
      int i0 = 0, i1 = 0;
#pragma unroll 8
      for (int e = 0; e < E_DIM; ++e) {
        float l = lrowp[e];
        if (l > v0) { v1 = v0; i1 = i0; v0 = l; i0 = e; }
        else if (l > v1) { v1 = l; i1 = e; }
      }
      float sum = 0.0f;
#pragma unroll 8
      for (int e = 0; e < E_DIM; ++e) sum += __expf(lrowp[e] - v0);
      const float inv = 1.0f / sum;
      const float p0 = inv;                     // exp(v0 - v0) / sum
      const float p1 = __expf(v1 - v0) * inv;
      const float rden = 1.0f / (p0 + p1 + EPS_GATE);
      out_cw[token * 2 + 0] = p0 * rden;
      out_cw[token * 2 + 1] = p1 * rden;
      out_idx[token * 2 + 0] = i0;
      out_idx[token * 2 + 1] = i1;
    }
  }
}

// ---------------------------------------------------------------------------
extern "C" void kernel_launch(void* const* d_in, const int* in_sizes, int n_in,
                              void* d_out, int out_size, void* d_ws, size_t ws_size,
                              hipStream_t stream) {
  const float* x = (const float*)d_in[0];        // [B,N,D] fp32
  const float* w = (const float*)d_in[1];        // [E,D]   fp32

  const int wN       = in_sizes[1];              // E*D
  const int n_tokens = in_sizes[0] / D_DIM;      // B*N

  unsigned short* whi = (unsigned short*)d_ws;   // E*D bf16 hi
  unsigned short* wlo = whi + wN;                // E*D bf16 lo  (512 KB total)

  split_w_kernel<<<(wN + 255) / 256, 256, 0, stream>>>(w, whi, wlo, wN);

  float* out_cw = (float*)d_out;                 // combine_weights [B,N,2]
  int*   out_idx = (int*)d_out + (size_t)n_tokens * 2;  // expert_index [B,N,2]

  const int nblocks = (n_tokens + 127) / 128;
  gating_kernel<<<nblocks, 256, 0, stream>>>(x, whi, wlo, out_cw, out_idx, n_tokens);
}
